// FlexLocalFrameAttention_46806553592108
// MI455X (gfx1250) — matrix operand order repaired and checked by reading, not compiled, not run on hardware
//
#include <hip/hip_runtime.h>
#include <hip/hip_bf16.h>

// ---------------------------------------------------------------------------
// Problem constants (from the reference)
// ---------------------------------------------------------------------------
#define BATCH   2
#define FRAMES  32
#define NTOK    128
#define DIM_    1024
#define HEADS   16
#define HD_     64
#define CHUNK_  4
#define CC      (FRAMES / CHUNK_)          // 8 chunks
#define LL      (CHUNK_ * NTOK)            // 512 rows per chunk
#define TT      (FRAMES * NTOK)            // 4096 tokens per batch
#define MROWS   (BATCH * TT)               // 8192 total rows
#define INNER   (HEADS * HD_)              // 1024
#define QKVN    (3 * INNER)                // 3072

typedef __attribute__((ext_vector_type(16))) __bf16 bf16x16;
typedef __attribute__((ext_vector_type(8)))  __bf16 bf16x8;
typedef __attribute__((ext_vector_type(2)))  __bf16 bf16x2;
typedef __attribute__((ext_vector_type(8)))  float  f32x8;
typedef __attribute__((ext_vector_type(4)))  float  f32x4;

#define WMMA_BF16(A, Bv, C) \
  __builtin_amdgcn_wmma_f32_16x16x32_bf16(false, (A), false, (Bv), (short)0, (C), false, false)

static __device__ __forceinline__ bf16x16 cat8(bf16x8 lo, bf16x8 hi) {
  return __builtin_shufflevector(lo, hi, 0,1,2,3,4,5,6,7,8,9,10,11,12,13,14,15);
}

// A-matrix fragment (16x32 bf16): lane holds row (lane&15); K elems 0-7 are
// K = kh*8..kh*8+7, elems 8-15 are K = 16+kh*8.. (kh = lane>>4).  Two b128 loads.
static __device__ __forceinline__ bf16x16 load_frag_a(const __bf16* base, int ldk, int lane) {
  const __bf16* p = base + (lane & 15) * ldk + ((lane >> 4) << 3);
  return cat8(*(const bf16x8*)p, *(const bf16x8*)(p + 16));
}
// Same, but the row pointer is already per-lane (Q fragments from global).
static __device__ __forceinline__ bf16x16 load_frag_a_row(const __bf16* prow, int lane) {
  const __bf16* p = prow + ((lane >> 4) << 3);
  return cat8(*(const bf16x8*)p, *(const bf16x8*)(p + 16));
}
// B-matrix fragment (32x16): lane holds col (lane&15); K elems 0-7 are
// K = kh*8.., elems 8-15 are K = 16+kh*8.. (matches A).  Source tile is stored [col][K] row-major.  Two b128 loads.
static __device__ __forceinline__ bf16x16 load_frag_b(const __bf16* base, int ldk, int lane) {
  const __bf16* p = base + (lane & 15) * ldk + ((lane >> 4) << 3);
  return cat8(*(const bf16x8*)p, *(const bf16x8*)(p + 16));
}

static __device__ __forceinline__ bf16x8 cvt8(f32x4 a, f32x4 b) {
  bf16x8 o;
#pragma unroll
  for (int i = 0; i < 4; ++i) { o[i] = (__bf16)a[i]; o[4 + i] = (__bf16)b[i]; }
  return o;
}

// ---------------------------------------------------------------------------
// DPP16 row-rotate butterfly reductions over each 16-lane half (a DPP "row"
// on wave32).  Rotation steps 1,2,4,8 give a full 16-lane all-reduce for
// commutative ops.  Runs on the VALU (co-executes with WMMA) instead of the
// LDS pipe that ds_bpermute-based __shfl_xor would occupy.
// ---------------------------------------------------------------------------
template <int N>
static __device__ __forceinline__ float dpp_row_ror(float x) {
  int i = __builtin_amdgcn_update_dpp(0, __float_as_int(x),
                                      0x120 | N, 0xF, 0xF, true);  // row_ror:N
  return __int_as_float(i);
}
static __device__ __forceinline__ float rowmax16(float v) {
  v = fmaxf(v, dpp_row_ror<1>(v));
  v = fmaxf(v, dpp_row_ror<2>(v));
  v = fmaxf(v, dpp_row_ror<4>(v));
  v = fmaxf(v, dpp_row_ror<8>(v));
  return v;
}
static __device__ __forceinline__ float rowsum16(float v) {
  v += dpp_row_ror<1>(v);
  v += dpp_row_ror<2>(v);
  v += dpp_row_ror<4>(v);
  v += dpp_row_ror<8>(v);
  return v;
}

// ---------------------------------------------------------------------------
// Tiled GEMM:  C[M x N] = A[M x K] * W[N x K]^T   (both row-major over K)
// A is f32 or bf16 (template), W is f32, C is f32 or bf16 (template).
// Block = 128x128, BK = 64, 256 threads = 8 waves, wave tile = 64x32.
// Double-buffered LDS (one barrier per K-slab) + global prefetch one slab
// ahead.  16 WMMAs per wave per slab.
// ---------------------------------------------------------------------------
template <bool A_BF16, bool C_F32>
__global__ __launch_bounds__(256) void gemm_xwt_wmma(const void* __restrict__ Ap,
                                                     const float* __restrict__ W,
                                                     void* __restrict__ Cp,
                                                     int M, int N, int K) {
  constexpr int BK = 64;
  constexpr int LD = BK + 8;                     // 72 elems = 144B row (16B aligned)
  __shared__ __bf16 As[2][128][LD];
  __shared__ __bf16 Ws[2][128][LD];

  const int tid = threadIdx.x;
  const int lane = tid & 31;
  const int wv = tid >> 5;
  const int waveM = (wv >> 2) * 64;
  const int waveN = (wv & 3) * 32;
  const long blockM = (long)blockIdx.y * 128;
  const long blockN = (long)blockIdx.x * 128;

  const int row = tid >> 1;                      // 0..127
  const int cb  = (tid & 1) * 32;                // 32-element column segment

  // Stage one 128x64 K-slab of A and W into LDS buffer `bufi` (f32 -> bf16).
  auto stage = [&](int k0, int bufi) {
    if constexpr (A_BF16) {
      const __bf16* g = (const __bf16*)Ap + (blockM + row) * (long)K + k0 + cb;
      __builtin_prefetch(g + BK, 0, 1);          // next slab -> global_prefetch_b8
      *(bf16x8*)&As[bufi][row][cb]      = *(const bf16x8*)g;
      *(bf16x8*)&As[bufi][row][cb + 8]  = *(const bf16x8*)(g + 8);
      *(bf16x8*)&As[bufi][row][cb + 16] = *(const bf16x8*)(g + 16);
      *(bf16x8*)&As[bufi][row][cb + 24] = *(const bf16x8*)(g + 24);
    } else {
      const float* g = (const float*)Ap + (blockM + row) * (long)K + k0 + cb;
      __builtin_prefetch(g + BK, 0, 1);
      f32x4 f0 = *(const f32x4*)g,        f1 = *(const f32x4*)(g + 4);
      f32x4 f2 = *(const f32x4*)(g + 8),  f3 = *(const f32x4*)(g + 12);
      f32x4 f4 = *(const f32x4*)(g + 16), f5 = *(const f32x4*)(g + 20);
      f32x4 f6 = *(const f32x4*)(g + 24), f7 = *(const f32x4*)(g + 28);
      *(bf16x8*)&As[bufi][row][cb]      = cvt8(f0, f1);
      *(bf16x8*)&As[bufi][row][cb + 8]  = cvt8(f2, f3);
      *(bf16x8*)&As[bufi][row][cb + 16] = cvt8(f4, f5);
      *(bf16x8*)&As[bufi][row][cb + 24] = cvt8(f6, f7);
    }
    {
      const float* g = W + (blockN + row) * (long)K + k0 + cb;
      __builtin_prefetch(g + BK, 0, 1);
      f32x4 f0 = *(const f32x4*)g,        f1 = *(const f32x4*)(g + 4);
      f32x4 f2 = *(const f32x4*)(g + 8),  f3 = *(const f32x4*)(g + 12);
      f32x4 f4 = *(const f32x4*)(g + 16), f5 = *(const f32x4*)(g + 20);
      f32x4 f6 = *(const f32x4*)(g + 24), f7 = *(const f32x4*)(g + 28);
      *(bf16x8*)&Ws[bufi][row][cb]      = cvt8(f0, f1);
      *(bf16x8*)&Ws[bufi][row][cb + 8]  = cvt8(f2, f3);
      *(bf16x8*)&Ws[bufi][row][cb + 16] = cvt8(f4, f5);
      *(bf16x8*)&Ws[bufi][row][cb + 24] = cvt8(f6, f7);
    }
  };

  f32x8 acc[4][2] = {};

  stage(0, 0);
  int bufi = 0;
  for (int k0 = 0; k0 < K; k0 += BK) {
    __syncthreads();                             // buffer `bufi` fully staged
    if (k0 + BK < K) stage(k0 + BK, bufi ^ 1);   // overlap next slab with compute
#pragma unroll
    for (int ks = 0; ks < 2; ++ks) {
      bf16x16 af[4], bfr[2];
#pragma unroll
      for (int mi = 0; mi < 4; ++mi)
        af[mi] = load_frag_a(&As[bufi][waveM + mi * 16][ks * 32], LD, lane);
#pragma unroll
      for (int ni = 0; ni < 2; ++ni)
        bfr[ni] = load_frag_b(&Ws[bufi][waveN + ni * 16][ks * 32], LD, lane);
#pragma unroll
      for (int mi = 0; mi < 4; ++mi)
#pragma unroll
        for (int ni = 0; ni < 2; ++ni)
          acc[mi][ni] = WMMA_BF16(af[mi], bfr[ni], acc[mi][ni]);
    }
    bufi ^= 1;
  }

  // ---- writeback: C layout lane holds rows r + 8*(lane>>4), col lane&15 ----
  const int kh = lane >> 4, ln = lane & 15;
#pragma unroll
  for (int mi = 0; mi < 4; ++mi)
#pragma unroll
    for (int ni = 0; ni < 2; ++ni)
#pragma unroll
      for (int r = 0; r < 8; ++r) {
        long gM = blockM + waveM + mi * 16 + r + 8 * kh;
        long gN = blockN + waveN + ni * 16 + ln;
        if constexpr (C_F32) ((float*)Cp)[gM * N + gN] = acc[mi][ni][r];
        else                 ((__bf16*)Cp)[gM * N + gN] = (__bf16)acc[mi][ni][r];
      }
}

// ---------------------------------------------------------------------------
// Flash-style chunked local attention.
// Block = (q-tile of 64 rows) x chunk x (b,h);  4 waves, each owns 16 q rows.
// qkv layout per token: [q(1024) | k(1024) | v(1024)], head h at offset h*64.
// kv position kv in [0,2L): token = (c-1)*L + kv   (chunk 0 starts at kv=L).
// Softmax scale folded into the bf16 Q fragments; row reductions done with
// DPP16 row-rotate butterflies on the VALU (LDS pipe stays free for WMMA
// fragment traffic).
// ---------------------------------------------------------------------------
__global__ __launch_bounds__(128) void attn_wmma(const __bf16* __restrict__ qkv,
                                                 __bf16* __restrict__ attn) {
  constexpr int LDK = HD_ + 8;                   // 72 elems = 144B (16B aligned)
  __shared__ __bf16 Ks[64][LDK];                 // K tile, row-major [kv][hd]
  __shared__ __bf16 Vt[64][LDK];                 // V tile transposed [hd][kv]
  __shared__ __bf16 Ps[4][16][LDK];              // per-wave P scratch [qrow][kv]

  const int tid = threadIdx.x, lane = tid & 31, wv = tid >> 5;
  const int b = blockIdx.z >> 4, h = blockIdx.z & 15;
  const int c = blockIdx.y, qt = blockIdx.x;
  const int kh = lane >> 4, ln = lane & 15;

  // Q fragments live in registers for the whole kernel (HD=64 -> 2 k-steps),
  // pre-scaled by 1/sqrt(HD) = 0.125 (exact power of two in bf16).
  const int qrow = qt * 64 + wv * 16 + ln;
  const __bf16* qp = qkv + ((size_t)(b * TT + c * LL + qrow)) * QKVN + h * HD_;
  bf16x16 aq0 = load_frag_a_row(qp, lane);
  bf16x16 aq1 = load_frag_a_row(qp + 32, lane);
#pragma unroll
  for (int i = 0; i < 16; ++i) {
    aq0[i] = (__bf16)((float)aq0[i] * 0.125f);
    aq1[i] = (__bf16)((float)aq1[i] * 0.125f);
  }

  float m[8], s[8];
  f32x8 o[4] = {};
#pragma unroll
  for (int r = 0; r < 8; ++r) { m[r] = -1e30f; s[r] = 0.0f; }

  // Staging roles: K rows -> 2 threads per kv row (32 contiguous bf16 each);
  // V transpose -> each thread owns a kv-row *pair* x 16 hd, so transposed
  // writes pack two bf16 into one ds_store_b32.
  const int rowk = tid >> 1, segk = tid & 1;
  const int rp = tid >> 2, hs = tid & 3;
  const int kvStart = (c == 0) ? LL : 0;

  for (int kv0 = kvStart; kv0 < 2 * LL; kv0 += 64) {
    __syncthreads();                             // prior tile fully consumed
    // ---- stage K (row-major) ----
    const int tk = (c - 1) * LL + kv0 + rowk;
    const __bf16* kp = qkv + ((size_t)(b * TT + tk)) * QKVN + INNER + h * HD_ + segk * 32;
    __builtin_prefetch(kp + (size_t)64 * QKVN, 0, 1);   // next kv tile
    bf16x8 k0 = *(const bf16x8*)kp,        k1 = *(const bf16x8*)(kp + 8);
    bf16x8 k2 = *(const bf16x8*)(kp + 16), k3 = *(const bf16x8*)(kp + 24);
    *(bf16x8*)&Ks[rowk][segk * 32]      = k0;
    *(bf16x8*)&Ks[rowk][segk * 32 + 8]  = k1;
    *(bf16x8*)&Ks[rowk][segk * 32 + 16] = k2;
    *(bf16x8*)&Ks[rowk][segk * 32 + 24] = k3;
    // ---- stage V transposed (row pair -> packed b32 stores) ----
    const int tv = (c - 1) * LL + kv0 + 2 * rp;
    const __bf16* vp0 = qkv + ((size_t)(b * TT + tv)) * QKVN + 2 * INNER + h * HD_ + hs * 16;
    const __bf16* vp1 = vp0 + QKVN;
    __builtin_prefetch(vp0 + (size_t)64 * QKVN, 0, 1);
    bf16x8 a0 = *(const bf16x8*)vp0, a1 = *(const bf16x8*)(vp0 + 8);
    bf16x8 b0 = *(const bf16x8*)vp1, b1 = *(const bf16x8*)(vp1 + 8);
#pragma unroll
    for (int j = 0; j < 8; ++j) {
      bf16x2 t0 = {a0[j], b0[j]};
      *(bf16x2*)&Vt[hs * 16 + j][2 * rp] = t0;
      bf16x2 t1 = {a1[j], b1[j]};
      *(bf16x2*)&Vt[hs * 16 + 8 + j][2 * rp] = t1;
    }
    __syncthreads();

    // ---- S = (Q/8) K^T : 16 q rows x 64 kv, 8 WMMAs ----
    f32x8 sc[4];
#pragma unroll
    for (int ct = 0; ct < 4; ++ct) {
      bf16x16 bk0 = load_frag_b(&Ks[ct * 16][0], LDK, lane);   // HD 0..31
      bf16x16 bk1 = load_frag_b(&Ks[ct * 16][32], LDK, lane);  // HD 32..63
      f32x8 z = {};
      z = WMMA_BF16(aq0, bk0, z);
      sc[ct] = WMMA_BF16(aq1, bk1, z);
    }

    // ---- online softmax (rows = r + 8*kh; DPP16 butterflies per half) ----
#pragma unroll
    for (int r = 0; r < 8; ++r) {
      float v = fmaxf(fmaxf(sc[0][r], sc[1][r]), fmaxf(sc[2][r], sc[3][r]));
      v = rowmax16(v);
      float mn = fmaxf(m[r], v);
      float alpha = __expf(m[r] - mn);
      m[r] = mn;
      s[r] = s[r] * alpha;
#pragma unroll
      for (int g = 0; g < 4; ++g) o[g][r] = o[g][r] * alpha;
    }
    float lsum[8];
#pragma unroll
    for (int r = 0; r < 8; ++r) lsum[r] = 0.0f;
#pragma unroll
    for (int ct = 0; ct < 4; ++ct)
#pragma unroll
      for (int r = 0; r < 8; ++r) {
        float p = __expf(sc[ct][r] - m[r]);
        lsum[r] += p;
        Ps[wv][r + 8 * kh][ct * 16 + ln] = (__bf16)p;   // C-layout -> A-layout via LDS
      }
#pragma unroll
    for (int r = 0; r < 8; ++r) s[r] += rowsum16(lsum[r]);

    // ---- O += P V : 8 WMMAs ----
#pragma unroll
    for (int ks = 0; ks < 2; ++ks) {
      bf16x16 ap = load_frag_a(&Ps[wv][0][ks * 32], LDK, lane);
#pragma unroll
      for (int g = 0; g < 4; ++g) {
        bf16x16 bv = load_frag_b(&Vt[g * 16][ks * 32], LDK, lane);
        o[g] = WMMA_BF16(ap, bv, o[g]);
      }
    }
  }

  // ---- epilogue: O / rowsum -> attn workspace (bf16, token-major) ----
#pragma unroll
  for (int r = 0; r < 8; ++r) {
    float inv = 1.0f / s[r];
    int tq = c * LL + qt * 64 + wv * 16 + r + 8 * kh;
    __bf16* op = attn + ((size_t)(b * TT + tq)) * INNER + h * HD_;
#pragma unroll
    for (int g = 0; g < 4; ++g)
      op[g * 16 + ln] = (__bf16)(o[g][r] * inv);
  }
}

// ---------------------------------------------------------------------------
// Launcher.  Workspace: [ qkv bf16 : 8192*3072 ][ attn bf16 : 8192*1024 ]
// ---------------------------------------------------------------------------
extern "C" void kernel_launch(void* const* d_in, const int* in_sizes, int n_in,
                              void* d_out, int out_size, void* d_ws, size_t ws_size,
                              hipStream_t stream) {
  (void)in_sizes; (void)n_in; (void)out_size; (void)ws_size;
  const float* x     = (const float*)d_in[0];   // [B,F,N,DIM] = [8192,1024]
  const float* w_qkv = (const float*)d_in[1];   // [3072,1024]
  const float* w_out = (const float*)d_in[2];   // [1024,1024]
  float* out = (float*)d_out;                   // [8192,1024]

  __bf16* qkv_ws  = (__bf16*)d_ws;
  __bf16* attn_ws = (__bf16*)((char*)d_ws + (size_t)MROWS * QKVN * sizeof(__bf16));

  // 1) qkv = x @ w_qkv^T   (f32 in, bf16 out)
  gemm_xwt_wmma<false, false>
      <<<dim3(QKVN / 128, MROWS / 128), 256, 0, stream>>>(x, w_qkv, qkv_ws,
                                                          MROWS, QKVN, DIM_);
  // 2) chunked local attention (bf16 in, bf16 out)
  attn_wmma<<<dim3(LL / 64, CC, BATCH * HEADS), 128, 0, stream>>>(qkv_ws, attn_ws);

  // 3) out = attn @ w_out^T  (bf16 A, f32 out)
  gemm_xwt_wmma<true, true>
      <<<dim3(INNER / 128, MROWS / 128), 256, 0, stream>>>(attn_ws, w_out, out,
                                                           MROWS, INNER, DIM_);
}
